// ImprovedGraphSAGE_88905823027737
// MI455X (gfx1250) — matrix operand order repaired and checked by reading, not compiled, not run on hardware
//
#include <hip/hip_runtime.h>
#include <math.h>

#define N_NODES 50000
#define N_EDGES 800000
#define D 128
#define M_TILES 3125   // 50000 / 16
#define LN_EPS 1e-5f

typedef __attribute__((ext_vector_type(16))) __bf16 bf16x16;
typedef __attribute__((ext_vector_type(4)))  __bf16 bf16x4;
typedef __attribute__((ext_vector_type(8)))  float  floatx8;
typedef __attribute__((ext_vector_type(8)))  int    intx8;

// ---------- helpers ----------
__device__ __forceinline__ float bf2f(unsigned s) {
    return __uint_as_float(s << 16);
}
__device__ __forceinline__ void atomAddF(float* p, float v) {
    __hip_atomic_fetch_add(p, v, __ATOMIC_RELAXED, __HIP_MEMORY_SCOPE_AGENT);
}

__device__ __forceinline__ floatx8 wmma_bf16(intx8 a, intx8 b, floatx8 c) {
    return __builtin_amdgcn_wmma_f32_16x16x32_bf16(
        false, __builtin_bit_cast(bf16x16, a),
        false, __builtin_bit_cast(bf16x16, b),
        (short)0, c, false, false);
}

// 16 bf16 from two contiguous 16-byte runs -> fragment VGPRs 0..7
__device__ __forceinline__ intx8 load2x16B(const __bf16* p0, const __bf16* p1) {
    int4 lo = *(const int4*)p0;
    int4 hi = *(const int4*)p1;
    intx8 f;
    f[0] = lo.x; f[1] = lo.y; f[2] = lo.z; f[3] = lo.w;
    f[4] = hi.x; f[5] = hi.y; f[6] = hi.z; f[7] = hi.w;
    return f;
}

// A fragment (16x32 bf16): lane row m, K runs {kb+8h..+7} and {kb+16+8h..+7}
__device__ __forceinline__ intx8 load_a(const __bf16* mat, int m, int kb, int half) {
    const __bf16* row = mat + m * D + kb + half * 8;
    return load2x16B(row, row + 16);
}

// B fragment (32x16 bf16) from N-major (transposed) weights: K runs {kb+koff/2..+7} and {kb+16+koff/2..+7}
__device__ __forceinline__ intx8 load_b(const __bf16* wt, int n, int kb, int koff) {
    const __bf16* p = wt + n * D + kb + koff / 2;
    return load2x16B(p, p + 16);
}

// ---------- setup kernels ----------
__global__ void k_zero(float* p, int n) {
    int i = blockIdx.x * blockDim.x + threadIdx.x;
    if (i < n) p[i] = 0.0f;
}

__global__ void k_degree(const int* dst, float* cnt) {
    int e = blockIdx.x * blockDim.x + threadIdx.x;
    if (e < N_EDGES) atomAddF(cnt + dst[e], 1.0f);
}

__global__ void k_invcnt(float* cnt) {
    int i = blockIdx.x * blockDim.x + threadIdx.x;
    if (i < N_NODES) cnt[i] = 1.0f / fmaxf(cnt[i], 1.0f);
}

// transpose + convert all 7 weight matrices (K-major fp32 -> N-major bf16)
__global__ void k_wt(const float* __restrict__ Win, const float* __restrict__ Wl,
                     const float* __restrict__ Wr, __bf16* __restrict__ wt) {
    int idx = blockIdx.x * blockDim.x + threadIdx.x;
    if (idx >= 7 * D * D) return;
    int mat = idx >> 14;          // /16384
    int r   = idx & 16383;
    int n = r >> 7, k = r & 127;
    const float* W = (mat == 0) ? Win
                   : (mat <= 3) ? (Wl + (mat - 1) * D * D)
                                : (Wr + (mat - 4) * D * D);
    wt[mat * D * D + n * D + k] = (__bf16)W[k * D + n];
}

// fp32 -> bf16, 4 elements per thread (hardware cvt)
__global__ void k_cvt(const float* __restrict__ in, __bf16* __restrict__ out, int n4) {
    int i = blockIdx.x * blockDim.x + threadIdx.x;
    if (i >= n4) return;
    float4 v = ((const float4*)in)[i];
    bf16x4 o;
    o[0] = (__bf16)v.x; o[1] = (__bf16)v.y; o[2] = (__bf16)v.z; o[3] = (__bf16)v.w;
    ((bf16x4*)out)[i] = o;
}

// agg * inv_cnt[row] -> bf16 (mean aggregation folded here)
__global__ void k_cvt_scaled(const float* __restrict__ in, const float* __restrict__ rowscale,
                             __bf16* __restrict__ out) {
    int i = blockIdx.x * blockDim.x + threadIdx.x;   // 4 elements each
    if (i >= N_NODES * D / 4) return;
    float s = rowscale[i >> 5];                      // (i*4)/128
    float4 v = ((const float4*)in)[i];
    bf16x4 o;
    o[0] = (__bf16)(v.x * s); o[1] = (__bf16)(v.y * s);
    o[2] = (__bf16)(v.z * s); o[3] = (__bf16)(v.w * s);
    ((bf16x4*)out)[i] = o;
}

// ---------- GEMM kernels ----------
// h = relu(xbf @ W_in + b_in); also bf16 mirror. One wave per 16x16 tile, B hoisted.
__global__ void k_gemm_in(const __bf16* __restrict__ xbf, const __bf16* __restrict__ wt,
                          const float* __restrict__ bin,
                          float* __restrict__ h, __bf16* __restrict__ hbf) {
    int wave = threadIdx.x >> 5;
    int lane = threadIdx.x & 31;
    int lr = lane & 15, half = lane >> 4;
    int n = wave * 16 + lr;
    int koff = half * 16;

    intx8 B[4];
#pragma unroll
    for (int c = 0; c < 4; ++c) B[c] = load_b(wt, n, c * 32, koff);
    float bias = bin[n];

    for (int mt = blockIdx.x; mt < M_TILES; mt += gridDim.x) {
        int mb = mt * 16;
        floatx8 acc = {};
#pragma unroll
        for (int c = 0; c < 4; ++c)
            acc = wmma_bf16(load_a(xbf, mb + lr, c * 32, half), B[c], acc);
#pragma unroll
        for (int r = 0; r < 8; ++r) {
            int m = mb + r + half * 8;
            float v = acc[r] + bias;
            v = v > 0.0f ? v : 0.0f;
            h[m * D + n] = v;
            hbf[m * D + n] = (__bf16)v;
        }
    }
}

// tmp = aggbf @ Wl + bl + hbf @ Wr   (dual WMMA chain, bias in C init, B hoisted)
__global__ void k_gemm_sage(const __bf16* __restrict__ aggbf, const __bf16* __restrict__ hbf,
                            const __bf16* __restrict__ wtl, const __bf16* __restrict__ wtr,
                            const float* __restrict__ bl, float* __restrict__ tmp) {
    int wave = threadIdx.x >> 5;
    int lane = threadIdx.x & 31;
    int lr = lane & 15, half = lane >> 4;
    int n = wave * 16 + lr;
    int koff = half * 16;

    intx8 BL[4], BR[4];
#pragma unroll
    for (int c = 0; c < 4; ++c) {
        BL[c] = load_b(wtl, n, c * 32, koff);
        BR[c] = load_b(wtr, n, c * 32, koff);
    }
    float bias = bl[n];

    for (int mt = blockIdx.x; mt < M_TILES; mt += gridDim.x) {
        int mb = mt * 16;
        floatx8 acc;
#pragma unroll
        for (int r = 0; r < 8; ++r) acc[r] = bias;
#pragma unroll
        for (int c = 0; c < 4; ++c) {
            acc = wmma_bf16(load_a(aggbf, mb + lr, c * 32, half), BL[c], acc);
            acc = wmma_bf16(load_a(hbf,   mb + lr, c * 32, half), BR[c], acc);
        }
#pragma unroll
        for (int r = 0; r < 8; ++r) {
            int m = mb + r + half * 8;
            tmp[m * D + n] = acc[r];
        }
    }
}

// ---------- message passing ----------
// one wave per edge: 32 lanes x 4 features; bf16 gather -> fp32 L2-resident atomics
__global__ void k_scatter(const __bf16* __restrict__ hbf,
                          const int* __restrict__ src, const int* __restrict__ dst,
                          float* __restrict__ agg) {
    int tid = blockIdx.x * blockDim.x + threadIdx.x;
    int e = tid >> 5;
    if (e >= N_EDGES) return;
    int lane = threadIdx.x & 31;
    int s = src[e];
    int d = dst[e];
    const unsigned* p = (const unsigned*)(hbf + s * D) + lane * 2;  // 4 bf16
    unsigned u0 = p[0], u1 = p[1];
    float* q = agg + d * D + lane * 4;
    atomAddF(q + 0, bf2f(u0 & 0xFFFFu));
    atomAddF(q + 1, bf2f(u0 >> 16));
    atomAddF(q + 2, bf2f(u1 & 0xFFFFu));
    atomAddF(q + 3, bf2f(u1 >> 16));
}

// ---------- LN + ReLU + residual ----------
__global__ void k_ln_relu_res(const float* __restrict__ tmp,
                              const float* __restrict__ gamma, const float* __restrict__ beta,
                              float* __restrict__ h, __bf16* __restrict__ hbf) {
    __shared__ float red_s[4];
    __shared__ float red_q[4];
    int m = blockIdx.x;
    int f = threadIdx.x;
    float v = tmp[m * D + f];
    float s = v, q = v * v;
#pragma unroll
    for (int o = 16; o > 0; o >>= 1) {
        s += __shfl_xor(s, o, 32);
        q += __shfl_xor(q, o, 32);
    }
    if ((f & 31) == 0) { red_s[f >> 5] = s; red_q[f >> 5] = q; }
    __syncthreads();
    float S = red_s[0] + red_s[1] + red_s[2] + red_s[3];
    float Q = red_q[0] + red_q[1] + red_q[2] + red_q[3];
    float mu = S * (1.0f / D);
    float var = Q * (1.0f / D) - mu * mu;
    float rstd = rsqrtf(var + LN_EPS);
    float o = (v - mu) * rstd * gamma[f] + beta[f];
    o = o > 0.0f ? o : 0.0f;
    float out = o + h[m * D + f];
    h[m * D + f] = out;
    hbf[m * D + f] = (__bf16)out;
}

// ---------- output head ----------
__global__ void k_out_head(const float* __restrict__ h, const float* __restrict__ Wout,
                           const float* __restrict__ bout, float* __restrict__ out) {
    int wid = (blockIdx.x * blockDim.x + threadIdx.x) >> 5;
    if (wid >= N_NODES) return;
    int lane = threadIdx.x & 31;
    const float4 hv = ((const float4*)(h + wid * D))[lane];
    int f = lane * 4;
    float d0 = hv.x * Wout[(f + 0) * 2 + 0] + hv.y * Wout[(f + 1) * 2 + 0] +
               hv.z * Wout[(f + 2) * 2 + 0] + hv.w * Wout[(f + 3) * 2 + 0];
    float d1 = hv.x * Wout[(f + 0) * 2 + 1] + hv.y * Wout[(f + 1) * 2 + 1] +
               hv.z * Wout[(f + 2) * 2 + 1] + hv.w * Wout[(f + 3) * 2 + 1];
#pragma unroll
    for (int o = 16; o > 0; o >>= 1) {
        d0 += __shfl_xor(d0, o, 32);
        d1 += __shfl_xor(d1, o, 32);
    }
    if (lane == 0) {
        float l0 = d0 + bout[0];
        float l1 = d1 + bout[1];
        float mx = fmaxf(l0, l1);
        float lse = mx + logf(expf(l0 - mx) + expf(l1 - mx));
        out[wid * 2 + 0] = l0 - lse;
        out[wid * 2 + 1] = l1 - lse;
    }
}

// ---------- launch ----------
extern "C" void kernel_launch(void* const* d_in, const int* in_sizes, int n_in,
                              void* d_out, int out_size, void* d_ws, size_t ws_size,
                              hipStream_t stream) {
    const float* x    = (const float*)d_in[0];
    const int*   ei   = (const int*)d_in[1];
    const float* Win  = (const float*)d_in[2];
    const float* bin  = (const float*)d_in[3];
    const float* Wl   = (const float*)d_in[4];
    const float* bl   = (const float*)d_in[5];
    const float* Wr   = (const float*)d_in[6];
    const float* gam  = (const float*)d_in[7];
    const float* bet  = (const float*)d_in[8];
    const float* Wout = (const float*)d_in[9];
    const float* bout = (const float*)d_in[10];
    float* out = (float*)d_out;

    const int* src = ei;
    const int* dst = ei + N_EDGES;

    char* ws = (char*)d_ws;
    float*  h     = (float*)(ws);                       // 25.6 MB
    __bf16* hbf   = (__bf16*)(ws + 25600000);           // 12.8 MB
    float*  agg   = (float*)(ws + 38400000);            // 25.6 MB (aliased by tmp)
    float*  tmp   = agg;                                // agg dead after k_cvt_scaled
    __bf16* xbf   = (__bf16*)(ws + 64000000);           // 12.8 MB (aliased by aggbf)
    __bf16* aggbf = xbf;                                // x dead after k_gemm_in
    float*  cnt   = (float*)(ws + 76800000);            // 0.2 MB
    __bf16* wt    = (__bf16*)(ws + 77000032 + 224);     // 0.23 MB (64B aligned: 77000256)

    // degree -> 1/max(deg,1)
    k_zero<<<(N_NODES + 255) / 256, 256, 0, stream>>>(cnt, N_NODES);
    k_degree<<<(N_EDGES + 255) / 256, 256, 0, stream>>>(dst, cnt);
    k_invcnt<<<(N_NODES + 255) / 256, 256, 0, stream>>>(cnt);

    // weights: transpose + bf16; x: bf16
    k_wt<<<(7 * D * D + 255) / 256, 256, 0, stream>>>(Win, Wl, Wr, wt);
    k_cvt<<<(N_NODES * D / 4 + 255) / 256, 256, 0, stream>>>(x, xbf, N_NODES * D / 4);

    // input projection
    k_gemm_in<<<1024, 256, 0, stream>>>(xbf, wt, bin, h, hbf);

    for (int i = 0; i < 3; ++i) {
        k_zero<<<(N_NODES * D + 255) / 256, 256, 0, stream>>>(agg, N_NODES * D);
        k_scatter<<<(N_EDGES * 32) / 256, 256, 0, stream>>>(hbf, src, dst, agg);
        k_cvt_scaled<<<(N_NODES * D / 4 + 255) / 256, 256, 0, stream>>>(agg, cnt, aggbf);
        k_gemm_sage<<<1024, 256, 0, stream>>>(aggbf, hbf,
                                              wt + (1 + i) * D * D, wt + (4 + i) * D * D,
                                              bl + i * D, tmp);
        k_ln_relu_res<<<N_NODES, D, 0, stream>>>(tmp, gam + i * D, bet + i * D, h, hbf);
    }

    k_out_head<<<(N_NODES * 32 + 255) / 256, 256, 0, stream>>>(h, Wout, bout, out);
}
